// InterpNetBase_79121887527504
// MI455X (gfx1250) — matrix operand order repaired and checked by reading, not compiled, not run on hardware
//
#include <hip/hip_runtime.h>

// ---------------------------------------------------------------------------
// InterpNet edge-MLP for MI455X (gfx1250, wave32, WMMA bf16).
// x = [latents[col] | pos_tgt[row]-pos_src[col] | dirs[col]]  (70 -> pad 96)
// h1 = relu(x@Win + b_in); h2 = relu(h1@W1 + b1); h3 = h2@W2 + b2
// val = h3@Wout + b_out ; segment-mean over row -> predictions (FILL if empty)
//
// Each wave owns a 16-edge WMMA tile and iterates TILES tiles, so the
// per-block weight staging (global->LDS bf16 transpose) is amortized 4x.
// Per-layer ordering of the wave-private activation tile uses wave_barrier
// (DS ops are in-order within a wave) instead of block-wide syncthreads.
// ---------------------------------------------------------------------------

typedef __attribute__((ext_vector_type(16))) __bf16 v16bf;
typedef __attribute__((ext_vector_type(8)))  float v8f;

union BF16x16 { v16bf v; unsigned short u[16]; };

#define WAVES 4         // waves per block
#define TILES 4         // 16-edge tiles per wave
#define IN_K 96         // 70 padded to 3 K-chunks of 32
#define FILL_VAL (-10000000.0f)

// float -> bf16 bits, round-to-nearest-even (bit-exact, no __bf16 scalar ops)
__device__ __forceinline__ unsigned short f2bf(float f) {
    union { float f; unsigned int u; } x; x.f = f;
    unsigned int r = x.u + 0x7FFFu + ((x.u >> 16) & 1u);
    return (unsigned short)(r >> 16);
}

__global__ void zero_ws_kernel(float* __restrict__ p, int n) {
    int i = blockIdx.x * blockDim.x + threadIdx.x;
    if (i < n) p[i] = 0.0f;
}

__global__ __launch_bounds__(WAVES * 32) void edge_mlp_kernel(
    const float* __restrict__ pos,      const float* __restrict__ latents,
    const float* __restrict__ dirs,     const float* __restrict__ posnm,
    const int*   __restrict__ row,      const int*   __restrict__ col,
    const float* __restrict__ Win,      const float* __restrict__ bin,
    const float* __restrict__ W1,       const float* __restrict__ b1,
    const float* __restrict__ W2,       const float* __restrict__ b2,
    const float* __restrict__ Wout,     const float* __restrict__ bout,
    float* __restrict__ sums, float* __restrict__ counts, int E)
{
    // Weights transposed in LDS: Wt[n][k] so B-fragments are contiguous in k.
    __shared__ __align__(16) unsigned short WinT[64 * IN_K];   // 12 KB
    __shared__ __align__(16) unsigned short W1T [64 * 64];     //  8 KB
    __shared__ __align__(16) unsigned short W2T [64 * 64];     //  8 KB
    __shared__ float Woutf[64], binf[64], b1f[64], b2f[64];
    // Per-wave activation tile, row-major [M=16][K=96] bf16 (reused per layer)
    __shared__ __align__(16) unsigned short xA[WAVES][16][IN_K]; // 12 KB

    const int tid = threadIdx.x;

    // ---- stage weights (bf16, transposed) ----
    for (int i = tid; i < IN_K * 64; i += WAVES * 32) {
        int k = i >> 6, n = i & 63;                   // Win is [70][64] row-major
        float v = (k < 70) ? Win[k * 64 + n] : 0.0f;
        WinT[n * IN_K + k] = f2bf(v);
    }
    for (int i = tid; i < 64 * 64; i += WAVES * 32) {
        int k = i >> 6, n = i & 63;
        W1T[n * 64 + k] = f2bf(W1[i]);
        W2T[n * 64 + k] = f2bf(W2[i]);
    }
    if (tid < 64) { Woutf[tid] = Wout[tid]; binf[tid] = bin[tid]; b1f[tid] = b1[tid]; b2f[tid] = b2[tid]; }
    __syncthreads();

    const int w    = tid >> 5;
    const int lane = tid & 31;
    const int mn   = lane & 15;     // A: row M ; B: col N ; C: N = mn
    const int h    = lane >> 4;     // lane half
    const int mg   = lane >> 1;     // gather: edge row handled by this lane pair
    const int sub  = lane & 1;      // gather: which half of the latent vector

    // Hoist per-lane bias / head weights out of the tile loop.
    float biasIn[4], bias1v[4], bias2v[4], woutv[4];
    #pragma unroll
    for (int nt = 0; nt < 4; ++nt) {
        biasIn[nt] = binf[nt * 16 + mn];
        bias1v[nt] = b1f [nt * 16 + mn];
        bias2v[nt] = b2f [nt * 16 + mn];
        woutv [nt] = Woutf[nt * 16 + mn];
    }
    const float boutv = bout[0];

    const long wavebase = ((long)blockIdx.x * WAVES + w) * (16 * TILES);

    // A-fragment: 16x32 bf16 (lanes 0-15: K 0-15, 16-31: K 16-31, contig in k; matches B)
    #define LOAD_AFRAG(dst, kc)                                              \
        {  _Pragma("unroll")                                                 \
           for (int j = 0; j < 16; ++j)                                      \
               (dst).u[j] = xA[w][mn][(kc) * 32 + h * 16 + j]; }
    // B-fragment from transposed weights: lanes 0-15: K 0-15, 16-31: K 16-31 (contig in k)
    #define LOAD_BFRAG(dst, Wt, ldk, kc, nt)                                 \
        {  _Pragma("unroll")                                                 \
           for (int j = 0; j < 16; ++j)                                      \
               (dst).u[j] = (Wt)[((nt) * 16 + mn) * (ldk) + (kc) * 32 + h * 16 + j]; }

    for (int t = 0; t < TILES; ++t) {
        const long base = wavebase + (long)t * 16;

        // ---- gather: build 16x96 bf16 input tile (2 lanes per edge row) ----
        {
            long e = base + mg;
            int ec = (e < E) ? (int)e : (E - 1);
            int c = col[ec];
            unsigned short* xrow = &xA[w][mg][0];
            const float4* lat4 = (const float4*)(latents + (size_t)c * 64 + sub * 32);
            #pragma unroll
            for (int j = 0; j < 8; ++j) {                 // 128-bit global loads
                float4 v = lat4[j];
                int o = sub * 32 + j * 4;
                xrow[o + 0] = f2bf(v.x); xrow[o + 1] = f2bf(v.y);
                xrow[o + 2] = f2bf(v.z); xrow[o + 3] = f2bf(v.w);
            }
            if (sub == 0) {
                int r = row[ec];
                float4 pt = ((const float4*)posnm)[r];
                float4 ps = ((const float4*)pos)[c];
                float4 dv = ((const float4*)dirs)[c];
                xrow[64] = f2bf(pt.y - ps.y);
                xrow[65] = f2bf(pt.z - ps.z);
                xrow[66] = f2bf(pt.w - ps.w);
                xrow[67] = f2bf(dv.y); xrow[68] = f2bf(dv.z); xrow[69] = f2bf(dv.w);
                #pragma unroll
                for (int j = 70; j < 80; ++j) xrow[j] = 0;
            } else {
                #pragma unroll
                for (int j = 80; j < 96; ++j) xrow[j] = 0;
            }
        }
        __builtin_amdgcn_wave_barrier();   // wave-local LDS ordering (DS in-order/wave)

        // ---------------- layer 1: x(16x96) @ Win(96x64) ----------------
        v8f acc[4];
        {
            BF16x16 a0, a1, a2;
            LOAD_AFRAG(a0, 0); LOAD_AFRAG(a1, 1); LOAD_AFRAG(a2, 2);
            #pragma unroll
            for (int nt = 0; nt < 4; ++nt) {
                v8f c = {};
                BF16x16 b;
                LOAD_BFRAG(b, WinT, IN_K, 0, nt);
                c = __builtin_amdgcn_wmma_f32_16x16x32_bf16(false, a0.v, false, b.v, (short)0, c, false, false);
                LOAD_BFRAG(b, WinT, IN_K, 1, nt);
                c = __builtin_amdgcn_wmma_f32_16x16x32_bf16(false, a1.v, false, b.v, (short)0, c, false, false);
                LOAD_BFRAG(b, WinT, IN_K, 2, nt);
                c = __builtin_amdgcn_wmma_f32_16x16x32_bf16(false, a2.v, false, b.v, (short)0, c, false, false);
                acc[nt] = c;
            }
        }
        __builtin_amdgcn_wave_barrier();
        #pragma unroll
        for (int nt = 0; nt < 4; ++nt)
            #pragma unroll
            for (int g = 0; g < 8; ++g) {              // C layout: M = g+8h, N = nt*16+mn
                float v = acc[nt][g] + biasIn[nt];
                xA[w][g + 8 * h][nt * 16 + mn] = f2bf(fmaxf(v, 0.0f));
            }
        __builtin_amdgcn_wave_barrier();

        // ---------------- layer 2: h1(16x64) @ W1(64x64) ----------------
        {
            BF16x16 a0, a1;
            LOAD_AFRAG(a0, 0); LOAD_AFRAG(a1, 1);
            #pragma unroll
            for (int nt = 0; nt < 4; ++nt) {
                v8f c = {};
                BF16x16 b;
                LOAD_BFRAG(b, W1T, 64, 0, nt);
                c = __builtin_amdgcn_wmma_f32_16x16x32_bf16(false, a0.v, false, b.v, (short)0, c, false, false);
                LOAD_BFRAG(b, W1T, 64, 1, nt);
                c = __builtin_amdgcn_wmma_f32_16x16x32_bf16(false, a1.v, false, b.v, (short)0, c, false, false);
                acc[nt] = c;
            }
        }
        __builtin_amdgcn_wave_barrier();
        #pragma unroll
        for (int nt = 0; nt < 4; ++nt)
            #pragma unroll
            for (int g = 0; g < 8; ++g) {
                float v = acc[nt][g] + bias1v[nt];
                xA[w][g + 8 * h][nt * 16 + mn] = f2bf(fmaxf(v, 0.0f));
            }
        __builtin_amdgcn_wave_barrier();

        // ---------------- layer 3: h2(16x64) @ W2(64x64) (no relu) -------
        {
            BF16x16 a0, a1;
            LOAD_AFRAG(a0, 0); LOAD_AFRAG(a1, 1);
            #pragma unroll
            for (int nt = 0; nt < 4; ++nt) {
                v8f c = {};
                BF16x16 b;
                LOAD_BFRAG(b, W2T, 64, 0, nt);
                c = __builtin_amdgcn_wmma_f32_16x16x32_bf16(false, a0.v, false, b.v, (short)0, c, false, false);
                LOAD_BFRAG(b, W2T, 64, 1, nt);
                c = __builtin_amdgcn_wmma_f32_16x16x32_bf16(false, a1.v, false, b.v, (short)0, c, false, false);
                acc[nt] = c;
            }
        }

        // ---------------- head: h3(16x64) @ Wout(64x1) + segment atomics --
        float p[8];
        #pragma unroll
        for (int g = 0; g < 8; ++g) {
            float s = 0.0f;
            #pragma unroll
            for (int nt = 0; nt < 4; ++nt)
                s += (acc[nt][g] + bias2v[nt]) * woutv[nt];
            #pragma unroll
            for (int off = 1; off < 16; off <<= 1)     // reduce within 16-lane half
                s += __shfl_xor(s, off, 32);
            p[g] = s;
        }
        if (mn < 8) {                                  // 16 lanes emit the 16 rows
            int m_out = 8 * h + mn;
            long e_out = base + m_out;
            if (e_out < E) {
                float val = 0.0f;
                #pragma unroll
                for (int g = 0; g < 8; ++g) val = (mn == g) ? p[g] : val;
                val += boutv;
                int tt = row[e_out];
                atomicAdd(&sums[tt], val);
                atomicAdd(&counts[tt], 1.0f);
            }
        }
        __builtin_amdgcn_wave_barrier();   // done with xA before next tile's gather
    }
}

__global__ void finalize_kernel(const float* __restrict__ sums,
                                const float* __restrict__ counts,
                                float* __restrict__ out, int n) {
    int i = blockIdx.x * blockDim.x + threadIdx.x;
    if (i < n) {
        float c = counts[i];
        out[i] = (c > 0.0f) ? (sums[i] / c) : FILL_VAL;
    }
}

extern "C" void kernel_launch(void* const* d_in, const int* in_sizes, int n_in,
                              void* d_out, int out_size, void* d_ws, size_t ws_size,
                              hipStream_t stream) {
    const float* pos     = (const float*)d_in[0];
    const float* latents = (const float*)d_in[1];
    const float* dirs    = (const float*)d_in[2];
    const float* posnm   = (const float*)d_in[3];
    const int*   row     = (const int*)d_in[4];
    const int*   col     = (const int*)d_in[5];
    const float* Win     = (const float*)d_in[6];
    const float* bin     = (const float*)d_in[7];
    const float* W1      = (const float*)d_in[8];
    const float* b1      = (const float*)d_in[9];
    const float* W2      = (const float*)d_in[10];
    const float* b2      = (const float*)d_in[11];
    const float* Wout    = (const float*)d_in[12];
    const float* bout    = (const float*)d_in[13];

    const int E     = in_sizes[4];
    const int n_tgt = out_size;

    float* sums   = (float*)d_ws;
    float* counts = sums + n_tgt;

    int zn = 2 * n_tgt;
    zero_ws_kernel<<<(zn + 255) / 256, 256, 0, stream>>>(sums, zn);

    const int edges_per_block = WAVES * TILES * 16;   // 256
    int blocks = (E + edges_per_block - 1) / edges_per_block;
    edge_mlp_kernel<<<blocks, WAVES * 32, 0, stream>>>(
        pos, latents, dirs, posnm, row, col,
        Win, bin, W1, b1, W2, b2, Wout, bout,
        sums, counts, E);

    finalize_kernel<<<(n_tgt + 255) / 256, 256, 0, stream>>>(sums, counts, (float*)d_out, n_tgt);
}
